// Actor_ObstacleEncoder_65695819760374
// MI455X (gfx1250) — matrix operand order repaired and checked by reading, not compiled, not run on hardware
//
#include <hip/hip_runtime.h>

// ---------------------------------------------------------------------------
// Types for CDNA5 WMMA (gfx1250, wave32)
// ---------------------------------------------------------------------------
typedef __bf16 bf16;
typedef __attribute__((ext_vector_type(16))) __bf16 v16bf;
typedef __attribute__((ext_vector_type(8)))  __bf16 v8bf;
typedef __attribute__((ext_vector_type(4)))  __bf16 v4bf;
typedef __attribute__((ext_vector_type(8)))  float  v8f;

#define HID     128
#define IN_DIM  96
#define NROWS   8192   // B * L
#define BATCH   1024
#define LMK     8
#define ICHUNK  4                      // i-blocks per LDS chunk
#define NCHUNK  (IN_DIM / ICHUNK)      // 24
#define IBLK_B  32768                  // bytes per full hw2 i-block (16384 bf16)
#define HBLK_B  16384                  // bytes per half i-block (one output half)

static __device__ __forceinline__ float fast_tanh(float x) {
#if __has_builtin(__builtin_amdgcn_tanhf)
  return __builtin_amdgcn_tanhf(x);          // v_tanh_f32 on gfx1250
#elif __has_builtin(__builtin_amdgcn_tanh_f32)
  return __builtin_amdgcn_tanh_f32(x);
#else
  return tanhf(x);
#endif
}

static __device__ __forceinline__ v8f wmma_bf16(v16bf a, v16bf b, v8f c) {
  // D(16x16,f32) = A(16x32,bf16) * B(32x16,bf16) + C
  return __builtin_amdgcn_wmma_f32_16x16x32_bf16(false, a, false, b, (short)0, c,
                                                 false, false);
}

// CDNA5 async global->LDS copy (16B per lane), tracked with ASYNCcnt.
static __device__ __forceinline__ void async_ld16(unsigned lds_off,
                                                  unsigned long long gaddr) {
  asm volatile("global_load_async_to_lds_b128 %0, %1, off"
               :: "v"(lds_off), "v"(gaddr)
               : "memory");
}

static __device__ __forceinline__ void wait_async0() {
#if __has_builtin(__builtin_amdgcn_s_wait_asynccnt)
  __builtin_amdgcn_s_wait_asynccnt(0);
#else
  asm volatile("s_wait_asynccnt 0x0" ::: "memory");
#endif
}

// A fragment (16x32 bf16) from row-major [rows, ld] storage.
static __device__ __forceinline__ v16bf load_a_frag(const bf16* __restrict__ base,
                                                    int ld, int row0, int k0,
                                                    int m, int half) {
  const bf16* p = base + (size_t)(row0 + m) * ld + k0 + half * 16;
  v16bf r;
#pragma unroll
  for (int t = 0; t < 8; ++t) { r[t] = p[t]; r[t + 8] = p[t + 8]; }
  return r;
}

// B fragment (32x16 bf16) from fragment-ordered storage (two 16B chunks/lane).
static __device__ __forceinline__ v16bf load_b_frag(const bf16* __restrict__ fp,
                                                    int lane) {
  const v8bf lo = *(const v8bf*)(fp + lane * 8);
  const v8bf hi = *(const v8bf*)(fp + 256 + lane * 8);
  v16bf r;
#pragma unroll
  for (int t = 0; t < 8; ++t) { r[t] = lo[t]; r[t + 8] = hi[t]; }
  return r;
}

// ---------------------------------------------------------------------------
// Weight conversion: f32 row-major [K, C] -> bf16 WMMA-B fragment order.
// B element (k, n): lane = (n&15) + 16*(k>=16), VGPR v = (k&15)>>1, h = k&1.
// Within fragment (512 elems): elem = chunk*256 + lane*8 + (v&3)*2 + h.
// ---------------------------------------------------------------------------
__global__ __launch_bounds__(256) void conv_frag_kernel(
    const float* __restrict__ src, bf16* __restrict__ dst, int K, int C) {
  const int tid = blockIdx.x * 256 + threadIdx.x;
  if (tid >= K * C) return;
  const int nks  = K >> 5;
  const int q    = tid & 7;
  const int lane = (tid >> 3) & 31;
  const int ch   = (tid >> 8) & 1;
  const int frag = tid >> 9;
  const int ks   = frag % nks;
  const int ct   = frag / nks;
  const int v    = ch * 4 + (q >> 1);
  const int h    = q & 1;
  const int k    = (ks << 5) + ((lane >> 4) << 4) + (v << 1) + h;
  const int c    = (ct << 4) + (lane & 15);
  dst[tid] = (bf16)src[(size_t)k * C + c];
}

// Build mlp_input x [N, 96] bf16: x[n] = [obs[b,0,0:64], obs[b,0,104+l*32 : +32]]
__global__ __launch_bounds__(256) void build_x_kernel(
    const float* __restrict__ obs, bf16* __restrict__ xbf) {
  const int tid = blockIdx.x * 256 + threadIdx.x;   // exactly N*96 threads
  const int n = tid / IN_DIM, i = tid % IN_DIM;
  const int b = n >> 3, l = n & 7;
  const float val = (i < 64) ? obs[b * 360 + i]
                             : obs[b * 360 + 104 + l * 32 + (i - 64)];
  xbf[tid] = (bf16)val;
}

// ---------------------------------------------------------------------------
// Generic fused GEMM: out = tanh(A @ B + bias), 16-row tile per wave.
// ---------------------------------------------------------------------------
template <int NKS1, int NKS2, bool A2MOD>
__global__ __launch_bounds__(256) void gemm_tanh_frag(
    const bf16* __restrict__ A1, const bf16* __restrict__ A2,
    const bf16* __restrict__ Bf, const float* __restrict__ bias,
    bf16* __restrict__ out_bf, float* __restrict__ out_f32) {
  constexpr int NKS = NKS1 + NKS2;
  const int w = threadIdx.x >> 5, lane = threadIdx.x & 31;
  const int m = lane & 15, half = lane >> 4;
  const int row0 = (blockIdx.x * 8 + w) * 16;

  v16bf afr[NKS];
#pragma unroll
  for (int ks = 0; ks < NKS1; ++ks)
    afr[ks] = load_a_frag(A1, NKS1 * 32, row0, ks * 32, m, half);
  if constexpr (NKS2 > 0) {
    const int row2 = A2MOD ? (row0 & (BATCH - 1)) : row0;
#pragma unroll
    for (int ks = 0; ks < NKS2; ++ks)
      afr[NKS1 + ks] = load_a_frag(A2, NKS2 * 32, row2, ks * 32, m, half);
  }

#pragma unroll
  for (int ct = 0; ct < 8; ++ct) {
    v8f acc;
#pragma unroll
    for (int r = 0; r < 8; ++r) acc[r] = 0.f;
#pragma unroll
    for (int ks = 0; ks < NKS; ++ks) {
      const v16bf b = load_b_frag(Bf + (size_t)(ct * NKS + ks) * 512, lane);
      acc = wmma_bf16(afr[ks], b, acc);
    }
    const float bn = bias[ct * 16 + m];
#pragma unroll
    for (int r = 0; r < 8; ++r) {
      const float t = fast_tanh(acc[r] + bn);
      const size_t idx = (size_t)(row0 + r + half * 8) * HID + ct * 16 + m;
      if (out_bf)  out_bf[idx] = (bf16)t;
      if (out_f32) out_f32[idx] = t;
    }
  }
}

// ---------------------------------------------------------------------------
// Fused hypernet-output + per-instance matvec (the 25.8 GFLOP core):
//   emb[n,o] = tanh( sum_i x[n,i] * tanh( h[n,:] . hw2[:, i*128+o] + hb2 ) )
// Two passes over the 128 output cols (4 ct each) keep live VGPRs < 256
// (acc 64 + A-frags 64 persistent).  Per pass: double-buffered async-to-LDS
// staging of 64 KB chunks (4 half-i-blocks); WG = 8 waves x 2 row-tiles.
// LDS: 2 x 64 KB = 128 KB.
// ---------------------------------------------------------------------------
__global__ __launch_bounds__(256) void hyper_emb_kernel(
    const bf16* __restrict__ h_bf, const bf16* __restrict__ x_bf,
    const bf16* __restrict__ hw2f, const float* __restrict__ hb2,
    float* __restrict__ emb_f32, bf16* __restrict__ emb_bf) {
  extern __shared__ unsigned char smem[];
  const unsigned lds_base = (unsigned)(unsigned long long)(size_t)smem;
  const int t = threadIdx.x;
  const int w = t >> 5, lane = t & 31;
  const int m = lane & 15, half = lane >> 4;
  const int r0 = blockIdx.x * 256 + w * 32;     // this wave's 32 rows

  // Preload A fragments (h rows, K=128 -> 4 k-steps, 2 row tiles) -- persistent
  v16bf afr[2][4];
#pragma unroll
  for (int rt = 0; rt < 2; ++rt)
#pragma unroll
    for (int ks = 0; ks < 4; ++ks)
      afr[rt][ks] = load_a_frag(h_bf, HID, r0 + rt * 16, ks * 32, m, half);

  for (int hp = 0; hp < 2; ++hp) {              // output-column half (ct 0-3 / 4-7)
    v8f acc[2][4];
#pragma unroll
    for (int rt = 0; rt < 2; ++rt)
#pragma unroll
      for (int ctl = 0; ctl < 4; ++ctl)
#pragma unroll
        for (int r = 0; r < 8; ++r) acc[rt][ctl][r] = 0.f;

    // Async-stage one 64 KB chunk: ICHUNK half-i-blocks of 16 KB each.
    auto stage = [&](int c, int buf) {
      const unsigned dst0 = lds_base + (unsigned)buf * (ICHUNK * HBLK_B);
      const unsigned long long src0 =
          (unsigned long long)(size_t)hw2f +
          (unsigned long long)c * (ICHUNK * IBLK_B) +
          (unsigned long long)hp * HBLK_B;
#pragma unroll
      for (int j = 0; j < 16; ++j) {            // 16 x 16B per thread
        const int ii = j >> 2, p = j & 3;
        const unsigned doff = (unsigned)(ii * HBLK_B + p * 4096 + t * 16);
        const unsigned long long soff =
            (unsigned long long)ii * IBLK_B + (unsigned)(p * 4096 + t * 16);
        async_ld16(dst0 + doff, src0 + soff);
      }
    };

    stage(0, 0);
    wait_async0();
    __syncthreads();

    for (int c = 0; c < NCHUNK; ++c) {
      const int buf = c & 1;
      if (c + 1 < NCHUNK) stage(c + 1, buf ^ 1);   // overlap copy with compute

      // xs scales for this chunk: 16 rows x ICHUNK i's, packed b64 loads
      v4bf xs[2][8];
#pragma unroll
      for (int rt = 0; rt < 2; ++rt)
#pragma unroll
        for (int r = 0; r < 8; ++r)
          xs[rt][r] = *(const v4bf*)(x_bf +
              (size_t)(r0 + rt * 16 + r + half * 8) * IN_DIM + c * ICHUNK);

      const bf16* lb = (const bf16*)(smem + (size_t)buf * (ICHUNK * HBLK_B));
#pragma unroll
      for (int ii = 0; ii < ICHUNK; ++ii) {
        const int i = c * ICHUNK + ii;
        float bn[4];                                // hoisted bias loads
#pragma unroll
        for (int ctl = 0; ctl < 4; ++ctl)
          bn[ctl] = hb2[i * HID + (hp * 4 + ctl) * 16 + m];
        const bf16* lbi = lb + ii * (HBLK_B / 2);   // elems
#pragma unroll
        for (int ctl = 0; ctl < 4; ++ctl) {
          const v16bf b0 = load_b_frag(lbi + (ctl * 4 + 0) * 512, lane);
          const v16bf b1 = load_b_frag(lbi + (ctl * 4 + 1) * 512, lane);
          const v16bf b2 = load_b_frag(lbi + (ctl * 4 + 2) * 512, lane);
          const v16bf b3 = load_b_frag(lbi + (ctl * 4 + 3) * 512, lane);
          v8f g0, g1;
#pragma unroll
          for (int r = 0; r < 8; ++r) { g0[r] = 0.f; g1[r] = 0.f; }
          g0 = wmma_bf16(afr[0][0], b0, g0); g1 = wmma_bf16(afr[1][0], b0, g1);
          g0 = wmma_bf16(afr[0][1], b1, g0); g1 = wmma_bf16(afr[1][1], b1, g1);
          g0 = wmma_bf16(afr[0][2], b2, g0); g1 = wmma_bf16(afr[1][2], b2, g1);
          g0 = wmma_bf16(afr[0][3], b3, g0); g1 = wmma_bf16(afr[1][3], b3, g1);
#pragma unroll
          for (int r = 0; r < 8; ++r) {
            acc[0][ctl][r] += fast_tanh(g0[r] + bn[ctl]) * (float)xs[0][r][ii];
            acc[1][ctl][r] += fast_tanh(g1[r] + bn[ctl]) * (float)xs[1][r][ii];
          }
        }
      }
      wait_async0();      // our chunk c+1 writes landed
      __syncthreads();    // everyone done reading buf & writing buf^1
    }

    // Epilogue for this column half: outer tanh + store
#pragma unroll
    for (int rt = 0; rt < 2; ++rt)
#pragma unroll
      for (int ctl = 0; ctl < 4; ++ctl)
#pragma unroll
        for (int r = 0; r < 8; ++r) {
          const float e = fast_tanh(acc[rt][ctl][r]);
          const size_t idx = (size_t)(r0 + rt * 16 + r + half * 8) * HID +
                             (hp * 4 + ctl) * 16 + m;
          emb_f32[idx] = e;
          emb_bf[idx]  = (bf16)e;
        }
  }
}

// Mean over landmarks: mean[b,o] = (1/8) sum_l emb[b*8+l, o]  -> bf16
__global__ __launch_bounds__(256) void mean_kernel(
    const float* __restrict__ emb_f32, bf16* __restrict__ mean_bf) {
  const int tid = blockIdx.x * 256 + threadIdx.x;   // BATCH*HID threads
  const int b = tid >> 7, o = tid & 127;
  float s = 0.f;
#pragma unroll
  for (int l = 0; l < LMK; ++l) s += emb_f32[(size_t)((b << 3) + l) * HID + o];
  mean_bf[tid] = (bf16)(s * 0.125f);
}

// logits[n] = a2[n,:] . aw3 + ab3
__global__ __launch_bounds__(256) void logits_kernel(
    const bf16* __restrict__ a2_bf, const float* __restrict__ aw3,
    const float* __restrict__ ab3, float* __restrict__ logits) {
  const int n = blockIdx.x * 256 + threadIdx.x;     // NROWS threads
  const bf16* row = a2_bf + (size_t)n * HID;
  float acc = ab3[0];
#pragma unroll 4
  for (int k = 0; k < HID; ++k) acc += (float)row[k] * aw3[k];
  logits[n] = acc;
}

// Softmax over L=8 per batch row + weighted combine of vals -> out [B,1,128]
__global__ __launch_bounds__(256) void softmax_combine_kernel(
    const float* __restrict__ logits, const float* __restrict__ vals,
    float* __restrict__ out) {
  const int w = threadIdx.x >> 5, lane = threadIdx.x & 31;
  const int b = blockIdx.x * 8 + w;
  float lg[LMK];
  float mx = -3.402823466e38f;
#pragma unroll
  for (int l = 0; l < LMK; ++l) { lg[l] = logits[b * LMK + l]; mx = fmaxf(mx, lg[l]); }
  float s = 0.f;
#pragma unroll
  for (int l = 0; l < LMK; ++l) { lg[l] = __expf(lg[l] - mx); s += lg[l]; }
  const float inv = 1.f / s;
  const int o0 = lane * 4;
  float a0 = 0.f, a1 = 0.f, a2 = 0.f, a3 = 0.f;
#pragma unroll
  for (int l = 0; l < LMK; ++l) {
    const float4 v = *(const float4*)(vals + (size_t)(b * LMK + l) * HID + o0);
    const float aa = lg[l] * inv;
    a0 += aa * v.x; a1 += aa * v.y; a2 += aa * v.z; a3 += aa * v.w;
  }
  float4 o; o.x = a0; o.y = a1; o.z = a2; o.w = a3;
  *(float4*)(out + (size_t)b * HID + o0) = o;
}

// ---------------------------------------------------------------------------
extern "C" void kernel_launch(void* const* d_in, const int* in_sizes, int n_in,
                              void* d_out, int out_size, void* d_ws, size_t ws_size,
                              hipStream_t stream) {
  (void)in_sizes; (void)n_in; (void)out_size; (void)ws_size;
  const float* obs = (const float*)d_in[0];
  const float* hw1 = (const float*)d_in[3];
  const float* hb1 = (const float*)d_in[4];
  const float* hw2 = (const float*)d_in[5];
  const float* hb2 = (const float*)d_in[6];
  const float* vw1 = (const float*)d_in[7];
  const float* vb1 = (const float*)d_in[8];
  const float* vw2 = (const float*)d_in[9];
  const float* vb2 = (const float*)d_in[10];
  const float* aw1 = (const float*)d_in[11];
  const float* ab1 = (const float*)d_in[12];
  const float* aw2 = (const float*)d_in[13];
  const float* ab2 = (const float*)d_in[14];
  const float* aw3 = (const float*)d_in[15];
  const float* ab3 = (const float*)d_in[16];
  float* out = (float*)d_out;

  // Workspace carve-out (~24.2 MB total), 256B aligned slices.
  char* ws = (char*)d_ws;
  size_t off = 0;
  auto take = [&](size_t bytes) -> char* {
    char* p = ws + off;
    off = (off + bytes + 255) & ~(size_t)255;
    return p;
  };
  bf16* hw1f    = (bf16*)take((size_t)IN_DIM * HID * 2);         // 24 KB
  bf16* hw2f    = (bf16*)take((size_t)HID * IN_DIM * HID * 2);   // 3.0 MB
  bf16* vw1f    = (bf16*)take((size_t)HID * HID * 2);
  bf16* vw2f    = (bf16*)take((size_t)HID * HID * 2);
  bf16* aw1f    = (bf16*)take((size_t)2 * HID * HID * 2);
  bf16* aw2f    = (bf16*)take((size_t)HID * HID * 2);
  bf16* x_bf    = (bf16*)take((size_t)NROWS * IN_DIM * 2);       // 1.5 MB
  bf16* h_bf    = (bf16*)take((size_t)NROWS * HID * 2);          // 2.0 MB
  float* embf32 = (float*)take((size_t)NROWS * HID * 4);         // 4.0 MB
  bf16* emb_bf  = (bf16*)take((size_t)NROWS * HID * 2);
  bf16* mean_bf = (bf16*)take((size_t)BATCH * HID * 2);
  bf16* v1_bf   = (bf16*)take((size_t)NROWS * HID * 2);
  float* valsf  = (float*)take((size_t)NROWS * HID * 4);         // 4.0 MB
  bf16* a1_bf   = (bf16*)take((size_t)NROWS * HID * 2);
  bf16* a2_bf   = (bf16*)take((size_t)NROWS * HID * 2);
  float* logits = (float*)take((size_t)NROWS * 4);

  // --- weight/input conversion to bf16 WMMA fragment order ---
  build_x_kernel<<<(NROWS * IN_DIM) / 256, 256, 0, stream>>>(obs, x_bf);
  conv_frag_kernel<<<(IN_DIM * HID + 255) / 256, 256, 0, stream>>>(hw1, hw1f, IN_DIM, HID);
  conv_frag_kernel<<<(HID * IN_DIM * HID + 255) / 256, 256, 0, stream>>>(hw2, hw2f, HID, IN_DIM * HID);
  conv_frag_kernel<<<(HID * HID + 255) / 256, 256, 0, stream>>>(vw1, vw1f, HID, HID);
  conv_frag_kernel<<<(HID * HID + 255) / 256, 256, 0, stream>>>(vw2, vw2f, HID, HID);
  conv_frag_kernel<<<(2 * HID * HID + 255) / 256, 256, 0, stream>>>(aw1, aw1f, 2 * HID, HID);
  conv_frag_kernel<<<(HID * HID + 255) / 256, 256, 0, stream>>>(aw2, aw2f, HID, HID);

  // --- hypernet layer 1: h = tanh(x @ hw1 + hb1)  (K=96 -> 3 k-steps) ---
  gemm_tanh_frag<3, 0, false><<<NROWS / 128, 256, 0, stream>>>(
      x_bf, nullptr, hw1f, hb1, h_bf, nullptr);

  // --- fused hypernet layer 2 + per-instance matvec -> emb ---
  hyper_emb_kernel<<<NROWS / 256, 256, 2 * ICHUNK * HBLK_B, stream>>>(
      h_bf, x_bf, hw2f, hb2, embf32, emb_bf);

  // --- mean pooling over landmarks ---
  mean_kernel<<<(BATCH * HID) / 256, 256, 0, stream>>>(embf32, mean_bf);

  // --- value MLP: vals = tanh(tanh(emb@vw1+vb1)@vw2+vb2) ---
  gemm_tanh_frag<4, 0, false><<<NROWS / 128, 256, 0, stream>>>(
      emb_bf, nullptr, vw1f, vb1, v1_bf, nullptr);
  gemm_tanh_frag<4, 0, false><<<NROWS / 128, 256, 0, stream>>>(
      v1_bf, nullptr, vw2f, vb2, nullptr, valsf);

  // --- attention MLP (concat handled as split-K; mean rows use n mod 1024) ---
  gemm_tanh_frag<4, 4, true><<<NROWS / 128, 256, 0, stream>>>(
      emb_bf, mean_bf, aw1f, ab1, a1_bf, nullptr);
  gemm_tanh_frag<4, 0, false><<<NROWS / 128, 256, 0, stream>>>(
      a1_bf, nullptr, aw2f, ab2, a2_bf, nullptr);
  logits_kernel<<<NROWS / 256, 256, 0, stream>>>(a2_bf, aw3, ab3, logits);

  // --- softmax over L=8 + weighted sum -> out [B,1,128] ---
  softmax_combine_kernel<<<BATCH / 8, 256, 0, stream>>>(logits, valsf, out);
}
